// Encoder_12541304505015
// MI455X (gfx1250) — matrix operand order repaired and checked by reading, not compiled, not run on hardware
//
#include <hip/hip_runtime.h>
#include <hip/hip_bf16.h>
#include <stdint.h>

// ---------------- problem constants ----------------
#define VOCAB 32000
#define EMB   1024
#define H     1024
#define FOURH 4096
#define BATCH 64
#define SEQ   512
#define MROWS (BATCH * SEQ)   // 32768 rows of the hoisted input GEMM

// ---------------- CDNA5 WMMA types ----------------
typedef __attribute__((ext_vector_type(16))) __bf16 v16bf;
typedef __attribute__((ext_vector_type(8)))  float  v8f;

union FragAB {
    v16bf v;
    uint4 q[2];
};

// ---------------- bf16 helpers (bit-level, RNE) ----------------
__device__ __forceinline__ uint16_t f32_to_bf16_bits(float f) {
    union { float f; uint32_t u; } x; x.f = f;
    uint32_t lsb = (x.u >> 16) & 1u;
    x.u += 0x7fffu + lsb;
    return (uint16_t)(x.u >> 16);
}
__device__ __forceinline__ float bf16_bits_to_f32(uint16_t b) {
    union { uint32_t u; float f; } x; x.u = ((uint32_t)b) << 16;
    return x.f;
}

// ---------------- fragment loaders (ISA 7.12.2 layouts) ----------------
// A: 16x32 bf16 tile from row-major [*, lda] matrix.
// lane L: m = L&15, half = L>>4
//   elems 0..7  <- A[m][k0 + 8*half     .. +7]
//   elems 8..15 <- A[m][k0 + 16 + 8*half .. +7]
__device__ __forceinline__ void load_a_frag(FragAB& fa, const uint16_t* __restrict__ A,
                                            int lda, int mBase, int k0) {
    int lane = threadIdx.x & 31;
    int m = lane & 15, half = lane >> 4;
    const uint16_t* p = A + (size_t)(mBase + m) * lda + k0;
    fa.q[0] = *(const uint4*)(p + 8 * half);
    fa.q[1] = *(const uint4*)(p + 16 + 8 * half);
}

// B: 32x16 bf16 tile where B[k][n] = W[n][k], W row-major [*, ldw].
// lane L: n = L&15, half = L>>4; elems 0..7: k = k0 + 8*half + e; elems 8..15: k = k0 + 16 + 8*half + (e-8)
__device__ __forceinline__ void load_b_frag(FragAB& fb, const uint16_t* __restrict__ W,
                                            int ldw, int nBase, int k0) {
    int lane = threadIdx.x & 31;
    int n = lane & 15, half = lane >> 4;
    const uint16_t* p = W + (size_t)(nBase + n) * ldw + k0;
    fb.q[0] = *(const uint4*)(p + 8 * half);
    fb.q[1] = *(const uint4*)(p + 16 + 8 * half);
}

// ---------------- setup kernels ----------------
__global__ void __launch_bounds__(256) k_f32_to_bf16(const float* __restrict__ src,
                                                     uint16_t* __restrict__ dst, int n) {
    int i = blockIdx.x * blockDim.x + threadIdx.x;
    if (i < n) dst[i] = f32_to_bf16_bits(src[i]);
}

__global__ void __launch_bounds__(256) k_init(const float* __restrict__ b_ih,
                                              const float* __restrict__ b_hh,
                                              float* __restrict__ bias,
                                              uint16_t* __restrict__ h0,
                                              float* __restrict__ c0) {
    int i = blockIdx.x * blockDim.x + threadIdx.x;
    if (i < FOURH) bias[i] = b_ih[i] + b_hh[i];
    if (i < BATCH * H) { h0[i] = 0; c0[i] = 0.0f; }
}

// Gather embedding rows into bf16, row m = s*BATCH + b (so per-step slabs are contiguous)
__global__ void __launch_bounds__(256) k_gather(const float* __restrict__ emb,
                                                const int* __restrict__ seq,
                                                uint16_t* __restrict__ X) {
    int row = blockIdx.x;                 // 0..MROWS-1
    int s = row >> 6, b = row & 63;
    int tok = seq[b * SEQ + s];
    const float* src = emb + (size_t)tok * EMB;
    uint16_t*    dst = X   + (size_t)row * EMB;
    for (int e = threadIdx.x; e < EMB; e += blockDim.x)
        dst[e] = f32_to_bf16_bits(src[e]);
}

// ---------------- hoisted input GEMM: PG = X @ W_ih^T + (b_ih+b_hh) ----------------
// Block tile 64(M) x 256(N), 8 waves; wave w: m-tile = w&3, n-group = w>>2 (8 n-tiles of 16)
__global__ void __launch_bounds__(256)
k_gemm_pregates(const uint16_t* __restrict__ X,    // [MROWS, EMB]  bf16
                const uint16_t* __restrict__ Wih,  // [FOURH, EMB]  bf16
                const float*    __restrict__ bias, // [FOURH]
                uint16_t*       __restrict__ PG) { // [MROWS, FOURH] bf16
    int wave = threadIdx.x >> 5;
    int lane = threadIdx.x & 31;
    int mBase  = blockIdx.x * 64 + (wave & 3) * 16;
    int nBase0 = blockIdx.y * 256 + (wave >> 2) * 128;

    v8f acc[8];
#pragma unroll
    for (int t = 0; t < 8; ++t)
#pragma unroll
        for (int r = 0; r < 8; ++r) acc[t][r] = 0.0f;

    for (int k0 = 0; k0 < EMB; k0 += 32) {
        FragAB fa;
        load_a_frag(fa, X, EMB, mBase, k0);
#pragma unroll
        for (int t = 0; t < 8; ++t) {
            FragAB fb;
            load_b_frag(fb, Wih, EMB, nBase0 + t * 16, k0);
            acc[t] = __builtin_amdgcn_wmma_f32_16x16x32_bf16(
                false, fa.v, false, fb.v, (short)0, acc[t], false, false);
        }
    }

    int half = lane >> 4, nl = lane & 15;
#pragma unroll
    for (int t = 0; t < 8; ++t) {
        int col = nBase0 + t * 16 + nl;
        float bv = bias[col];
#pragma unroll
        for (int r = 0; r < 8; ++r) {
            int row = mBase + 8 * half + r;
            PG[(size_t)row * FOURH + col] = f32_to_bf16_bits(acc[t][r] + bv);
        }
    }
}

// ---------------- one LSTM timestep ----------------
// Grid: 16 blocks; block owns hidden units [hid0, hid0+64) for all 64 batch rows,
// i.e. gate columns {g*1024 + hid0 + j : g in 0..3, j in 0..63} = 64x256 gates.
// GEMM part: gates += h_in @ W_hh^T ; then fused LSTM pointwise.
__global__ void __launch_bounds__(256)
k_lstm_step(const uint16_t* __restrict__ PGs,   // pre-gates this step [BATCH, FOURH] bf16
            const uint16_t* __restrict__ Whh,   // [FOURH, H] bf16 (L2-resident, 8MB)
            const uint16_t* __restrict__ h_in,  // [BATCH, H] bf16
            uint16_t*       __restrict__ h_out, // [BATCH, H] bf16
            float*          __restrict__ cst,   // [BATCH, H] f32 (in place)
            float*          __restrict__ hs,    // [BATCH, SEQ, H] f32 (encoder outputs)
            int s) {
    __shared__ float gsm[BATCH][256 + 4];

    int hid0 = blockIdx.x * 64;
    int wave = threadIdx.x >> 5;
    int lane = threadIdx.x & 31;
    int mBase = (wave & 3) * 16;   // batch rows
    int ngrp  = wave >> 2;         // 0/1 -> 8 local n-tiles each

    v8f acc[8];
#pragma unroll
    for (int t = 0; t < 8; ++t)
#pragma unroll
        for (int r = 0; r < 8; ++r) acc[t][r] = 0.0f;

    for (int k0 = 0; k0 < H; k0 += 32) {
        FragAB fa;
        load_a_frag(fa, h_in, H, mBase, k0);
#pragma unroll
        for (int t = 0; t < 8; ++t) {
            int lcBase = (ngrp * 8 + t) * 16;                 // 0..240 local gate col
            int nGlob  = (lcBase >> 6) * H + hid0 + (lcBase & 63);
            FragAB fb;
            load_b_frag(fb, Whh, H, nGlob, k0);
            acc[t] = __builtin_amdgcn_wmma_f32_16x16x32_bf16(
                false, fa.v, false, fb.v, (short)0, acc[t], false, false);
        }
    }

    // spill GEMM result to LDS (C layout: m = 8*half + r, n = lane&15)
    int half = lane >> 4, nl = lane & 15;
#pragma unroll
    for (int t = 0; t < 8; ++t) {
        int lc = (ngrp * 8 + t) * 16 + nl;
#pragma unroll
        for (int r = 0; r < 8; ++r)
            gsm[mBase + 8 * half + r][lc] = acc[t][r];
    }
    __syncthreads();

    // fused LSTM cell pointwise: thread -> column jj, 16 batch rows
    int jj = threadIdx.x & 63;
    int b0 = (threadIdx.x >> 6) * 16;
    int col = hid0 + jj;
    for (int bi = 0; bi < 16; ++bi) {
        int b = b0 + bi;
        const uint16_t* pg = PGs + (size_t)b * FOURH;
        float ir = gsm[b][jj]       + bf16_bits_to_f32(pg[0 * H + col]);
        float fr = gsm[b][64 + jj]  + bf16_bits_to_f32(pg[1 * H + col]);
        float gr = gsm[b][128 + jj] + bf16_bits_to_f32(pg[2 * H + col]);
        float orr= gsm[b][192 + jj] + bf16_bits_to_f32(pg[3 * H + col]);
        float ig = 1.0f / (1.0f + __expf(-ir));
        float fg = 1.0f / (1.0f + __expf(-fr));
        float gg = tanhf(gr);
        float og = 1.0f / (1.0f + __expf(-orr));
        size_t ci = (size_t)b * H + col;
        float cn = fg * cst[ci] + ig * gg;
        cst[ci] = cn;
        float hn = og * tanhf(cn);
        hs[(size_t)b * SEQ * H + (size_t)s * H + col] = hn;
        h_out[ci] = f32_to_bf16_bits(hn);
    }
}

// ---------------- finalize: h_last, c_last ----------------
__global__ void __launch_bounds__(256) k_final(const float* __restrict__ hs,
                                               const float* __restrict__ cst,
                                               float* __restrict__ h_last,
                                               float* __restrict__ c_last) {
    int i = blockIdx.x * blockDim.x + threadIdx.x;
    if (i < BATCH * H) {
        int b = i >> 10, j = i & 1023;
        h_last[i] = hs[(size_t)b * SEQ * H + (size_t)(SEQ - 1) * H + j];
        c_last[i] = cst[i];
    }
}

// ---------------- host side ----------------
extern "C" void kernel_launch(void* const* d_in, const int* in_sizes, int n_in,
                              void* d_out, int out_size, void* d_ws, size_t ws_size,
                              hipStream_t stream) {
    const float* emb  = (const float*)d_in[0];
    const float* W_ih = (const float*)d_in[1];
    const float* W_hh = (const float*)d_in[2];
    const float* b_ih = (const float*)d_in[3];
    const float* b_hh = (const float*)d_in[4];
    const int*   seq  = (const int*)d_in[5];

    float* out    = (float*)d_out;
    float* hs     = out;                                 // [B, S, H]
    float* h_last = out + (size_t)BATCH * SEQ * H;       // [B, H]
    float* c_last = h_last + (size_t)BATCH * H;          // [B, H]

    // workspace carve (256B aligned)
    char* ws = (char*)d_ws;
    size_t off = 0;
    auto carve = [&](size_t bytes) -> char* {
        char* p = ws + off;
        off += (bytes + 255) & ~(size_t)255;
        return p;
    };
    uint16_t* Wih16 = (uint16_t*)carve((size_t)FOURH * EMB * 2);    //  8 MB
    uint16_t* Whh16 = (uint16_t*)carve((size_t)FOURH * H * 2);      //  8 MB
    float*    bias  = (float*)   carve((size_t)FOURH * 4);
    uint16_t* X16   = (uint16_t*)carve((size_t)MROWS * EMB * 2);    // 64 MB
    uint16_t* PG    = (uint16_t*)carve((size_t)MROWS * FOURH * 2);  // 256 MB
    uint16_t* hbuf0 = (uint16_t*)carve((size_t)BATCH * H * 2);
    uint16_t* hbuf1 = (uint16_t*)carve((size_t)BATCH * H * 2);
    float*    cbuf  = (float*)   carve((size_t)BATCH * H * 4);
    (void)ws_size; (void)in_sizes; (void)n_in; (void)out_size;

    const int NW = FOURH * EMB;  // 4,194,304 weights per matrix
    k_f32_to_bf16<<<(NW + 255) / 256, 256, 0, stream>>>(W_ih, Wih16, NW);
    k_f32_to_bf16<<<(NW + 255) / 256, 256, 0, stream>>>(W_hh, Whh16, NW);
    k_init<<<(BATCH * H + 255) / 256, 256, 0, stream>>>(b_ih, b_hh, bias, hbuf0, cbuf);
    k_gather<<<MROWS, 256, 0, stream>>>(emb, seq, X16);

    dim3 gGemm(MROWS / 64, FOURH / 256);                 // 512 x 16 blocks
    k_gemm_pregates<<<gGemm, 256, 0, stream>>>(X16, Wih16, bias, PG);

    for (int s = 0; s < SEQ; ++s) {
        const uint16_t* hin = (s & 1) ? hbuf1 : hbuf0;
        uint16_t*       hout= (s & 1) ? hbuf0 : hbuf1;
        k_lstm_step<<<H / 64, 256, 0, stream>>>(
            PG + (size_t)s * BATCH * FOURH, Whh16, hin, hout, cbuf, hs, s);
    }

    k_final<<<(BATCH * H + 255) / 256, 256, 0, stream>>>(hs, cbuf, h_last, c_last);
}
